// BasicBlock_5617817223625
// MI455X (gfx1250) — 8-bit matrix operand order repaired and checked by reading, compiled, not run on hardware
//
#include <hip/hip_runtime.h>

typedef int v8i __attribute__((ext_vector_type(8)));
typedef int v4i_vs __attribute__((vector_size(16)));

#define N_    32
#define C_    256
#define H_    56
#define W_    56
#define HW_   (H_*W_)
#define NCHW_ (N_*C_*HW_)

#if defined(__has_builtin)
#  if __has_builtin(__builtin_amdgcn_global_load_async_to_lds_b128) && \
      __has_builtin(__builtin_amdgcn_s_wait_asynccnt)
#    define HAVE_ASYNC_LDS 1
#  endif
#endif

#ifdef HAVE_ASYNC_LDS
typedef __attribute__((address_space(1))) v4i_vs g_v4i;
typedef __attribute__((address_space(3))) v4i_vs l_v4i;
#endif

__global__ void zero_sums(long long* sums) {
    int t = threadIdx.x;
    if (t < 512) sums[t] = 0LL;
}

__global__ __launch_bounds__(256) void binarize_x(const float* __restrict__ x,
                                                  signed char* __restrict__ xs) {
    int idx = blockIdx.x * 256 + threadIdx.x;
    int w = idx % W_;
    int t = idx / W_;
    int h = t % H_;  t /= H_;
    int c = t & 255;
    int n = t >> 8;
    float v = x[idx];
    xs[((n*H_ + h)*W_ + w)*C_ + c] = (signed char)((v > 0.0f) - (v < 0.0f));
}

__global__ __launch_bounds__(256) void binarize_w(const float* __restrict__ w,
                                                  signed char* __restrict__ wsg) {
    int idx = blockIdx.x * 256 + threadIdx.x;
    int ci = idx & 255;
    int t2 = idx >> 8;
    int co = t2 & 255;
    int tp = t2 >> 8;
    float v = w[(co*C_ + ci)*9 + tp];
    wsg[idx] = (signed char)((v > 0.0f) - (v < 0.0f));
}

__global__ __launch_bounds__(256) void conv_wmma(const signed char* __restrict__ xs,
                                                 const signed char* __restrict__ wsg,
                                                 short* __restrict__ y,
                                                 long long* __restrict__ sums) {
    __shared__ int4 smem4[864];
    signed char* lds = (signed char*)smem4;

    const int tid  = threadIdx.x;
    const int lane = tid & 31;
    const int wave = tid >> 5;

    int bx = blockIdx.x;
    const int tw = bx & 3;  bx >>= 2;
    const int h  = bx % H_;
    const int n  = bx / H_;
    const int w0 = tw * 16;
    const int co0 = blockIdx.y * 128 + wave * 16;

    for (int i = tid; i < 864; i += 256) {
        const int lin = i << 4;
        const int c16 = lin & 255;
        const int rc  = lin >> 8;
        const int r   = rc / 18;
        const int col = rc % 18;
        const int hh  = h + r - 1;
        const int ww  = w0 + col - 1;
        const bool valid = ((unsigned)hh < (unsigned)H_) && ((unsigned)ww < (unsigned)W_);
#ifdef HAVE_ASYNC_LDS
        if (valid) {
            __builtin_amdgcn_global_load_async_to_lds_b128(
                (g_v4i*)(xs + (((n*H_ + hh)*W_ + ww) * C_ + c16)),
                (l_v4i*)(lds + lin),
                0, 0);
        } else {
            smem4[i] = make_int4(0, 0, 0, 0);
        }
#else
        int4 v = make_int4(0, 0, 0, 0);
        if (valid)
            v = *(const int4*)(xs + (((n*H_ + hh)*W_ + ww) * C_ + c16));
        smem4[i] = v;
#endif
    }
#ifdef HAVE_ASYNC_LDS
    __builtin_amdgcn_s_wait_asynccnt(0);
#endif
    __syncthreads();

    const int m     = lane & 15;
    const int kbase = (lane >> 4) << 4;
    const int px    = lane & 15;
    const int kb0   = (lane >> 4) << 4;

    v8i acc = {0, 0, 0, 0, 0, 0, 0, 0};

    for (int t = 0; t < 9; ++t) {
        const int kh = t / 3, kw = t % 3;
        const signed char* arow = wsg + ((t*C_) + co0 + m) * C_ + kbase;
        const int brow = ((kh * 18) + px + kw) << 8;
        for (int c0 = 0; c0 < C_; c0 += 64) {
            const int2* ap = (const int2*)(arow + c0);
            int2 p0 = ap[0], p1 = ap[1], p2 = ap[4], p3 = ap[5];
            v8i A = {p0.x, p0.y, p1.x, p1.y, p2.x, p2.y, p3.x, p3.y};
            const int4* bp = (const int4*)(lds + brow + c0 + kb0);
            int4 q0 = bp[0];
            int4 q1 = bp[2];
            v8i B = {q0.x, q0.y, q0.z, q0.w, q1.x, q1.y, q1.z, q1.w};
            acc = __builtin_amdgcn_wmma_i32_16x16x64_iu8(true, A, true, B, acc,
                                                         false, false);
        }
    }

    const int cobase = co0 + ((lane >> 4) << 3);
    const int pxw    = w0 + (lane & 15);
    const bool pvalid = (pxw < W_);

    if (pvalid) {
        #pragma unroll
        for (int g = 0; g < 8; ++g)
            y[(((n*C_) + cobase + g)*H_ + h)*W_ + pxw] = (short)acc[g];
    }

    #pragma unroll
    for (int g = 0; g < 8; ++g) {
        int s  = pvalid ? acc[g] : 0;
        int sq = s * s;
        #pragma unroll
        for (int msk = 8; msk >= 1; msk >>= 1) {
            s  += __shfl_xor(s,  msk, 32);
            sq += __shfl_xor(sq, msk, 32);
        }
        if ((lane & 15) == 0) {
            atomicAdd((unsigned long long*)&sums[cobase + g],
                      (unsigned long long)(long long)s);
            atomicAdd((unsigned long long*)&sums[256 + cobase + g],
                      (unsigned long long)(long long)sq);
        }
    }
}

__global__ __launch_bounds__(256) void finalize(const short* __restrict__ y,
                                                const float* __restrict__ x,
                                                const float* __restrict__ gamma,
                                                const float* __restrict__ beta,
                                                const long long* __restrict__ sums,
                                                float* __restrict__ out) {
    const int idx = blockIdx.x * 256 + threadIdx.x;
    const int c = (idx / HW_) & 255;
    const double inv_cnt = 1.0 / (double)(N_ * HW_);
    const double mean = (double)sums[c] * inv_cnt;
    const double var  = (double)sums[256 + c] * inv_cnt - mean * mean;
    const float r = rsqrtf((float)var + 1e-5f);
    const float yh = ((float)y[idx] - (float)mean) * r;
    out[idx] = gamma[c] * yh + beta[c] + x[idx];
}

extern "C" void kernel_launch(void* const* d_in, const int* in_sizes, int n_in,
                              void* d_out, int out_size, void* d_ws, size_t ws_size,
                              hipStream_t stream) {
    const float* x     = (const float*)d_in[0];
    const float* w     = (const float*)d_in[1];
    const float* gamma = (const float*)d_in[2];
    const float* beta  = (const float*)d_in[3];
    float* out = (float*)d_out;

    char* ws = (char*)d_ws;
    short*       y    = (short*)ws;
    signed char* xs   = (signed char*)(ws + 51380224);
    signed char* wsg  = (signed char*)(ws + 51380224 + 25690112);
    long long*   sums = (long long*)(ws + 51380224 + 25690112 + 589824);

    zero_sums <<<1,             512, 0, stream>>>(sums);
    binarize_x<<<NCHW_ / 256,   256, 0, stream>>>(x, xs);
    binarize_w<<<(9*C_*C_)/256, 256, 0, stream>>>(w, wsg);
    conv_wmma <<<dim3(N_*H_*4, 2), 256, 0, stream>>>(xs, wsg, y, sums);
    finalize  <<<NCHW_ / 256,   256, 0, stream>>>(y, x, gamma, beta, sums, out);
}
